// Encoder_9603546873856
// MI455X (gfx1250) — compile-verified, matrix operand order checked by reading, not run on hardware
//
#include <hip/hip_runtime.h>
#include <hip/hip_bf16.h>
#include <stdint.h>

#define B_   256
#define T_   64
#define N_   21
#define DIN_ 128
#define H_   256
#define M_   (B_*N_)   // 5376 rows

typedef __attribute__((ext_vector_type(16))) __bf16 v16bf;
typedef __attribute__((ext_vector_type(8)))  float  v8f;

union V16 { v16bf v; uint4 q[2]; };

__device__ __forceinline__ v8f v8f_zero() {
  v8f z = {0.f,0.f,0.f,0.f,0.f,0.f,0.f,0.f};
  return z;
}
__device__ __forceinline__ unsigned short f2bf(float f) {
  union { float f; unsigned u; } x; x.f = f;
  unsigned r = x.u + 0x7FFFu + ((x.u >> 16) & 1u);   // round-to-nearest-even
  return (unsigned short)(r >> 16);
}
__device__ __forceinline__ float sigf(float x) { return 1.0f / (1.0f + __expf(-x)); }

// ---------------------------------------------------------------------------
// Prep: L1-normalize the 4 graphs, keep rowsum(Gn) for bias folding, b_ih+b_hh.
// meta layout (floats): [g*462 .. ] gn(441)+rs(21) for g in {h1,h2,lstm,fc}; bcomb @1848 (1024)
// ---------------------------------------------------------------------------
__global__ void prep_meta(const float* Gh1, const float* Gh2, const float* Gl, const float* Gfc,
                          const float* bih, const float* bhh, float* meta) {
  int t = threadIdx.x;
  if (t < 84) {
    int g = t / 21, i = t % 21;
    const float* G = (g == 0) ? Gh1 : (g == 1) ? Gh2 : (g == 2) ? Gl : Gfc;
    float* gnp = meta + g * 462;
    float* rsp = gnp + 441;
    float sa = 0.f, s = 0.f;
    for (int j = 0; j < 21; ++j) { float v = G[i*21+j]; sa += fabsf(v); s += v; }
    sa = fmaxf(sa, 1e-12f);
    float inv = 1.0f / sa;
    for (int j = 0; j < 21; ++j) gnp[i*21+j] = G[i*21+j] * inv;
    rsp[i] = s * inv;   // rowsum of normalized G -> scales the (node-constant) bias
  }
  float* bcomb = meta + 1848;
  for (int i = t; i < 1024; i += blockDim.x) bcomb[i] = bih[i] + bhh[i];
}

// ---------------------------------------------------------------------------
// Convert W [K,Nn] f32 row-major -> bf16 tiles in the WMMA B-register layout:
// tile (kc, nt): 32 lanes x 16 bf16 contiguous per lane (2x b128 at load time).
// ---------------------------------------------------------------------------
__global__ void swizzle_w(const float* __restrict__ W, unsigned short* __restrict__ dst,
                          int K, int Nn) {
  int idx = blockIdx.x * blockDim.x + threadIdx.x;
  if (idx >= K * Nn) return;
  int slot = idx & 15;
  int lane = (idx >> 4) & 31;
  int tile = idx >> 9;
  int ntiles = Nn >> 4;
  int nt = tile % ntiles;
  int kc = tile / ntiles;
  int n  = nt * 16 + (lane & 15);
  int h  = lane >> 4;
  int k  = kc * 32 + (slot < 8 ? h * 8 + slot : 16 + h * 8 + (slot - 8));
  dst[idx] = f2bf(W[(size_t)k * Nn + n]);
}

// ---------------------------------------------------------------------------
// Graph premix: dst[b,i,f] = sum_j Gn[i,j] * src[b,j,f], cast to bf16 (row-major).
// Exploits G@(XW) = (G@X)W so mixing happens on the NARROW operand.
// ---------------------------------------------------------------------------
__global__ void premix(const float* __restrict__ src, long sbat, int F,
                       const float* __restrict__ Gn, unsigned short* __restrict__ dst) {
  __shared__ float sG[441];
  for (int i = threadIdx.x; i < 441; i += blockDim.x) sG[i] = Gn[i];
  __syncthreads();
  int idx = blockIdx.x * blockDim.x + threadIdx.x;
  int b = idx / F, f = idx - b * F;
  const float* p = src + (size_t)b * sbat + f;
  float s[21];
  #pragma unroll
  for (int j = 0; j < 21; ++j) s[j] = p[(size_t)j * F];
  for (int i = 0; i < 21; ++i) {
    float acc = 0.f;
    #pragma unroll
    for (int j = 0; j < 21; ++j) acc = fmaf(sG[i*21 + j], s[j], acc);
    dst[((size_t)b * 21 + i) * F + f] = f2bf(acc);
  }
}

// ---------------------------------------------------------------------------
// Generic WMMA GEMM, Nn=256: out[M,256] = A[M,K](bf16) * Bsw + rs[row%21]*bias[col]
// act=1 applies tanh. Block = 16 rows x 256 cols, 8 waves x 2 accum tiles.
// (Called 3x total — not perf-critical.)
// ---------------------------------------------------------------------------
__global__ void gemm_n256(const unsigned short* __restrict__ A,
                          const unsigned short* __restrict__ Bsw,
                          const float* __restrict__ bias,
                          const float* __restrict__ rs,
                          int K, float* __restrict__ out, int act) {
  int tid  = threadIdx.x;
  int wave = tid >> 5, lane = tid & 31;
  int m = lane & 15, half = lane >> 4;
  int mbase = blockIdx.x * 16;
  int row = mbase + m;
  const unsigned short* arow = A + (size_t)row * K;
  v8f a0 = v8f_zero(), a1 = v8f_zero();
  int nt0 = wave * 2;
  int kchunks = K >> 5;
  for (int kc = 0; kc < kchunks; ++kc) {
    V16 af;
    af.q[0] = *(const uint4*)(arow + kc * 32 + half * 8);
    af.q[1] = *(const uint4*)(arow + kc * 32 + 16 + half * 8);
    const unsigned short* bp = Bsw + ((size_t)(kc * 16 + nt0) * 512) + lane * 16;
    __builtin_prefetch(arow + (kc + 1) * 32, 0, 3);
    V16 b0f, b1f;
    b0f.q[0] = *(const uint4*)(bp);
    b0f.q[1] = *(const uint4*)(bp + 8);
    b1f.q[0] = *(const uint4*)(bp + 512);
    b1f.q[1] = *(const uint4*)(bp + 520);
    a0 = __builtin_amdgcn_wmma_f32_16x16x32_bf16(false, af.v, false, b0f.v, (short)0, a0, false, false);
    a1 = __builtin_amdgcn_wmma_f32_16x16x32_bf16(false, af.v, false, b1f.v, (short)0, a1, false, false);
  }
  #pragma unroll
  for (int j = 0; j < 2; ++j) {
    v8f acc = j ? a1 : a0;
    int col = (nt0 + j) * 16 + m;             // lane%16 indexes N in C/D layout
    float bb = bias[col];
    #pragma unroll
    for (int v = 0; v < 8; ++v) {
      int rw = mbase + v + 8 * half;          // VGPR v: M=v (lanes<16) / 8+v (lanes>=16)
      float val = acc[v] + rs[rw % 21] * bb;
      if (act) val = tanhf(val);
      out[(size_t)rw * 256 + col] = val;
    }
  }
}

// ---------------------------------------------------------------------------
// Fused LSTM step, 32-row tile: each B fragment feeds TWO row-subtiles
// (halves the W_ih/W_hh L2 traffic vs a 16-row tile). 192 WMMAs per wave.
// gates staged through 128 KB LDS so i/f/g/o recombine for the c/h update.
// ---------------------------------------------------------------------------
__global__ void lstm_step(const unsigned short* __restrict__ xm,
                          const unsigned short* __restrict__ hm,
                          const unsigned short* __restrict__ Wih,
                          const unsigned short* __restrict__ Whh,
                          const float* __restrict__ bcomb,
                          const float* __restrict__ rs,
                          float* __restrict__ c, float* __restrict__ h) {
  __shared__ float gbuf[32 * 1024];           // 128 KB of the 320 KB WGP LDS
  int tid  = threadIdx.x;
  int wave = tid >> 5, lane = tid & 31;
  int m = lane & 15, half = lane >> 4;
  int mbase = blockIdx.x * 32;
  int row0 = mbase + m;                        // row subtile 0
  int row1 = mbase + 16 + m;                   // row subtile 1

  v8f acc0[8], acc1[8];
  #pragma unroll
  for (int j = 0; j < 8; ++j) { acc0[j] = v8f_zero(); acc1[j] = v8f_zero(); }

  {   // K = 128 over pre-mixed x_t
    const unsigned short* ar0 = xm + (size_t)row0 * 128;
    const unsigned short* ar1 = xm + (size_t)row1 * 128;
    for (int kc = 0; kc < 4; ++kc) {
      V16 a0f, a1f;
      a0f.q[0] = *(const uint4*)(ar0 + kc * 32 + half * 8);
      a0f.q[1] = *(const uint4*)(ar0 + kc * 32 + 16 + half * 8);
      a1f.q[0] = *(const uint4*)(ar1 + kc * 32 + half * 8);
      a1f.q[1] = *(const uint4*)(ar1 + kc * 32 + 16 + half * 8);
      __builtin_prefetch(ar0 + (kc + 1) * 32, 0, 3);
      __builtin_prefetch(ar1 + (kc + 1) * 32, 0, 3);
      #pragma unroll
      for (int j = 0; j < 8; ++j) {
        const unsigned short* bp = Wih + ((size_t)(kc * 64 + wave * 8 + j) * 512) + lane * 16;
        V16 bf_;
        bf_.q[0] = *(const uint4*)bp;
        bf_.q[1] = *(const uint4*)(bp + 8);
        acc0[j] = __builtin_amdgcn_wmma_f32_16x16x32_bf16(false, a0f.v, false, bf_.v, (short)0, acc0[j], false, false);
        acc1[j] = __builtin_amdgcn_wmma_f32_16x16x32_bf16(false, a1f.v, false, bf_.v, (short)0, acc1[j], false, false);
      }
    }
  }
  {   // K = 256 over pre-mixed h
    const unsigned short* ar0 = hm + (size_t)row0 * 256;
    const unsigned short* ar1 = hm + (size_t)row1 * 256;
    for (int kc = 0; kc < 8; ++kc) {
      V16 a0f, a1f;
      a0f.q[0] = *(const uint4*)(ar0 + kc * 32 + half * 8);
      a0f.q[1] = *(const uint4*)(ar0 + kc * 32 + 16 + half * 8);
      a1f.q[0] = *(const uint4*)(ar1 + kc * 32 + half * 8);
      a1f.q[1] = *(const uint4*)(ar1 + kc * 32 + 16 + half * 8);
      __builtin_prefetch(ar0 + (kc + 1) * 32, 0, 3);
      __builtin_prefetch(ar1 + (kc + 1) * 32, 0, 3);
      #pragma unroll
      for (int j = 0; j < 8; ++j) {
        const unsigned short* bp = Whh + ((size_t)(kc * 64 + wave * 8 + j) * 512) + lane * 16;
        V16 bf_;
        bf_.q[0] = *(const uint4*)bp;
        bf_.q[1] = *(const uint4*)(bp + 8);
        acc0[j] = __builtin_amdgcn_wmma_f32_16x16x32_bf16(false, a0f.v, false, bf_.v, (short)0, acc0[j], false, false);
        acc1[j] = __builtin_amdgcn_wmma_f32_16x16x32_bf16(false, a1f.v, false, bf_.v, (short)0, acc1[j], false, false);
      }
    }
  }

  // stage gates (+ folded bias) to LDS
  #pragma unroll
  for (int j = 0; j < 8; ++j) {
    int col = wave * 128 + j * 16 + m;
    float bb = bcomb[col];
    #pragma unroll
    for (int v = 0; v < 8; ++v) {
      int rl = v + 8 * half;
      gbuf[rl * 1024 + col]        = acc0[j][v] + rs[(mbase + rl) % 21] * bb;
      gbuf[(16 + rl) * 1024 + col] = acc1[j][v] + rs[(mbase + 16 + rl) % 21] * bb;
    }
  }
  __syncthreads();

  // LSTM elementwise update: 32 rows x 256 hidden
  for (int it = 0; it < 32; ++it) {
    int idx = it * 256 + tid;
    int r = idx >> 8, hc = idx & 255;
    float gi = gbuf[r * 1024 + hc];
    float gf = gbuf[r * 1024 + 256 + hc];
    float gg = gbuf[r * 1024 + 512 + hc];
    float go = gbuf[r * 1024 + 768 + hc];
    size_t p = (size_t)(mbase + r) * 256 + hc;
    float cp = c[p];
    float cn = sigf(gf) * cp + sigf(gi) * tanhf(gg);
    c[p] = cn;
    h[p] = sigf(go) * tanhf(cn);
  }
}

__global__ void copy_f32(const float* __restrict__ s, float* __restrict__ d, int n) {
  int i = blockIdx.x * blockDim.x + threadIdx.x;
  if (i < n) d[i] = s[i];
}

// ---------------------------------------------------------------------------
extern "C" void kernel_launch(void* const* d_in, const int* in_sizes, int n_in,
                              void* d_out, int out_size, void* d_ws, size_t ws_size,
                              hipStream_t stream) {
  const float* x    = (const float*)d_in[0];
  const float* Gh1  = (const float*)d_in[1];
  const float* Wh1  = (const float*)d_in[2];
  const float* bh1  = (const float*)d_in[3];
  const float* Gh2  = (const float*)d_in[4];
  const float* Wh2  = (const float*)d_in[5];
  const float* bh2  = (const float*)d_in[6];
  const float* Gl   = (const float*)d_in[7];
  const float* Wih  = (const float*)d_in[8];
  const float* bih  = (const float*)d_in[9];
  const float* Whh  = (const float*)d_in[10];
  const float* bhh  = (const float*)d_in[11];
  const float* Gfc  = (const float*)d_in[12];
  const float* Wfc  = (const float*)d_in[13];
  const float* bfc  = (const float*)d_in[14];

  char* ws = (char*)d_ws;
  float* meta  = (float*)ws;                          // 16 KB
  float* gn_h1 = meta + 0,    *rs_h1 = meta + 441;
  float* gn_h2 = meta + 462,  *rs_h2 = meta + 903;
  float* gn_l  = meta + 924,  *rs_l  = meta + 1365;
  float* gn_fc = meta + 1386, *rs_fc = meta + 1827;
  float* bcomb = meta + 1848;
  unsigned short* wih_sw = (unsigned short*)(ws + 16384);     // 256 KB
  unsigned short* whh_sw = (unsigned short*)(ws + 278528);    // 512 KB
  unsigned short* wh1_sw = (unsigned short*)(ws + 802816);    // 64 KB
  unsigned short* wh2_sw = (unsigned short*)(ws + 868352);    // 64 KB
  unsigned short* wfc_sw = (unsigned short*)(ws + 933888);    // 128 KB
  unsigned short* x0m1   = (unsigned short*)(ws + 1064960);   // 1.31 MB
  unsigned short* x0m2   = (unsigned short*)(ws + 2441216);   // 1.31 MB
  unsigned short* xm     = (unsigned short*)(ws + 3817472);   // 1.31 MB
  unsigned short* hm     = (unsigned short*)(ws + 5193728);   // 2.63 MB
  float* hbuf = (float*)(ws + 7946240);                       // 5.25 MB
  float* cbuf = (float*)(ws + 13451264);                      // 5.25 MB  (~18 MB total)

  prep_meta<<<1, 256, 0, stream>>>(Gh1, Gh2, Gl, Gfc, bih, bhh, meta);
  swizzle_w<<<(128 * 1024) / 256, 256, 0, stream>>>(Wih, wih_sw, 128, 1024);
  swizzle_w<<<(256 * 1024) / 256, 256, 0, stream>>>(Whh, whh_sw, 256, 1024);
  swizzle_w<<<(128 * 256) / 256, 256, 0, stream>>>(Wh1, wh1_sw, 128, 256);
  swizzle_w<<<(128 * 256) / 256, 256, 0, stream>>>(Wh2, wh2_sw, 128, 256);
  swizzle_w<<<(256 * 256) / 256, 256, 0, stream>>>(Wfc, wfc_sw, 256, 256);

  const long xbat = (long)T_ * N_ * DIN_;   // batch stride inside x[B,T,N,DIN]

  // h0 = (Gh1@x0) W_h1 + rs*b ; c0 likewise
  premix<<<(B_ * DIN_) / 256, 256, 0, stream>>>(x, xbat, DIN_, gn_h1, x0m1);
  premix<<<(B_ * DIN_) / 256, 256, 0, stream>>>(x, xbat, DIN_, gn_h2, x0m2);
  gemm_n256<<<M_ / 16, 256, 0, stream>>>(x0m1, wh1_sw, bh1, rs_h1, 128, hbuf, 0);
  gemm_n256<<<M_ / 16, 256, 0, stream>>>(x0m2, wh2_sw, bh2, rs_h2, 128, cbuf, 0);

  // recurrence (stream order provides the inter-step barrier)
  for (int t = 0; t < T_; ++t) {
    premix<<<(B_ * DIN_) / 256, 256, 0, stream>>>(x + (size_t)t * N_ * DIN_, xbat, DIN_, gn_l, xm);
    premix<<<(B_ * H_) / 256, 256, 0, stream>>>(hbuf, (long)N_ * H_, H_, gn_l, hm);
    lstm_step<<<M_ / 32, 256, 0, stream>>>(xm, hm, wih_sw, whh_sw, bcomb, rs_l, cbuf, hbuf);
  }

  // out = tanh((Gfc@h_T) W_fc + rs*b_fc) ; y_last = h_T
  premix<<<(B_ * H_) / 256, 256, 0, stream>>>(hbuf, (long)N_ * H_, H_, gn_fc, hm);
  gemm_n256<<<M_ / 16, 256, 0, stream>>>(hm, wfc_sw, bfc, rs_fc, 256, (float*)d_out, 1);
  copy_f32<<<(M_ * H_ + 255) / 256, 256, 0, stream>>>(hbuf, (float*)d_out + (size_t)M_ * H_, M_ * H_);
}
